// GraphConstruct_69063074119735
// MI455X (gfx1250) — matrix operand order repaired and checked by reading, not compiled, not run on hardware
//
#include <hip/hip_runtime.h>
#include <cstdint>
#include <cstddef>

// ---------------- types for WMMA (gfx1250, wave32) ----------------
typedef __attribute__((ext_vector_type(8)))  unsigned short u16x8;
typedef __attribute__((ext_vector_type(16))) unsigned short u16x16;
typedef __attribute__((ext_vector_type(16))) __bf16         v16bf;
typedef __attribute__((ext_vector_type(8)))  float          v8f;

#define HH 64
#define WW 64
#define CC 64
#define M1 62               // patches per side
#define NPATCH 3844         // 62*62
#define EDIM 576            // c * 3 * 3
#define KTOP 5
#define WINS 30
#define BATCH 2
#define INFF 3.0e38f

static __device__ __forceinline__ int imin(int a, int b) { return a < b ? a : b; }
static __device__ __forceinline__ int imax(int a, int b) { return a > b ? a : b; }

// float -> bf16 (round to nearest even)
static __device__ __forceinline__ unsigned short f2bf(float f) {
    unsigned u = __float_as_uint(f);
    unsigned r = ((u >> 16) & 1u) + 0x7FFFu;
    return (unsigned short)((u + r) >> 16);
}

static __device__ __forceinline__ void ins5(float d, int n, float td[KTOP], int ti[KTOP]) {
    if (d < td[4]) {
        td[4] = d; ti[4] = n;
#pragma unroll
        for (int s = 4; s > 0; --s) {
            if (td[s] < td[s - 1]) {
                float f = td[s]; td[s] = td[s - 1]; td[s - 1] = f;
                int   ii = ti[s]; ti[s] = ti[s - 1]; ti[s - 1] = ii;
            }
        }
    }
}

// ------------------------------------------------------------------
// Kernel 1: build bf16 patch-feature matrices + f32 squared norms.
// grid (3844, b, 2), block 576. z==0 -> x tensor, z==1 -> y tensor.
// ------------------------------------------------------------------
__global__ void prep_kernel(const float* __restrict__ xe, const float* __restrict__ ye,
                            unsigned short* __restrict__ xfb, unsigned short* __restrict__ yfb,
                            float* __restrict__ sx, float* __restrict__ sy)
{
    __shared__ float red[576];
    const int p = blockIdx.x;
    const int b = blockIdx.y;
    const int which = blockIdx.z;
    const int e = threadIdx.x;
    const int c = e / 9, pp = e % 9, pi = pp / 3, pj = pp % 3;
    const int i = p / M1, j = p % M1;

    const float* src = which ? ye : xe;
    unsigned short* dst = which ? yfb : xfb;
    float* nrm = which ? sy : sx;

    const float v = src[(((size_t)b * CC + c) * HH + (i + pi)) * WW + (j + pj)];
    dst[(size_t)(b * NPATCH + p) * EDIM + e] = f2bf(v);

    red[e] = v * v;
    __syncthreads();
    if (e < 64) red[e] += red[e + 512];
    __syncthreads();
    for (int s = 256; s > 0; s >>= 1) {
        if (e < s) red[e] += red[e + s];
        __syncthreads();
    }
    if (e == 0) nrm[b * NPATCH + p] = red[0];
}

// ------------------------------------------------------------------
// Kernel 2: windowed distances via WMMA bf16 + streaming top-5 +
// per-channel L1 sums for the selected matches.
// grid (4 query-col tiles, 62 query rows, b), block 256 (8 waves).
// ------------------------------------------------------------------
__launch_bounds__(256, 1)
__global__ void nlm_topk_kernel(const unsigned short* __restrict__ xfb,
                                const unsigned short* __restrict__ yfb,
                                const float* __restrict__ sx, const float* __restrict__ sy,
                                const float* __restrict__ xe, const float* __restrict__ ye,
                                int* __restrict__ idx_out, float* __restrict__ sv_ws,
                                float* __restrict__ dsum_ws)
{
    __shared__ float dots[8][16][17];        // per-wave 16x16 dot tile (padded)
    __shared__ float pd[16][16][KTOP];       // partial top-5 dists  [query][list]
    __shared__ int   pidx[16][16][KTOP];     // partial top-5 indices
    __shared__ int   fi_s[16][KTOP];         // final winning indices

    const int jt = blockIdx.x, qi = blockIdx.y, b = blockIdx.z;
    const int j0 = jt * 16;
    const int si   = imin(imax(qi - 15, 0), 32);   // window row start (same for all queries in tile)
    const int col0 = imin(imax(j0 - 15, 0), 32);   // union-window col start

    const int tid  = threadIdx.x;
    const int wave = tid >> 5, lane = tid & 31;
    const int half = lane >> 4;                    // 0: lanes 0-15, 1: lanes 16-31
    const int qsl  = lane & 15;                    // M row / N col slot

    // ---- preload all 18 A-fragments (queries) into registers ----
    const int aq  = imin(j0 + qsl, M1 - 1);
    const unsigned short* ap = yfb + (size_t)(b * NPATCH + qi * M1 + aq) * EDIM;
    const int akoff = half ? 8 : 0;                // A: K {0..7,16..23} | {8..15,24..31}
    v16bf A[18];
#pragma unroll
    for (int kc = 0; kc < 18; ++kc) {
        u16x8 lo = *reinterpret_cast<const u16x8*>(ap + kc * 32 + akoff);
        u16x8 hi = *reinterpret_cast<const u16x8*>(ap + kc * 32 + akoff + 16);
        u16x16 m = __builtin_shufflevector(lo, hi, 0,1,2,3,4,5,6,7,8,9,10,11,12,13,14,15);
        union { u16x16 u; v16bf bf; } cv; cv.u = m;
        A[kc] = cv.bf;
    }

    float tp[KTOP]; int ip[KTOP];
#pragma unroll
    for (int k = 0; k < KTOP; ++k) { tp[k] = INFF; ip[k] = 0; }

    const int bkoff = half ? 8 : 0;                // B: K {0..7,16..23} | {8..15,24..31}
    const int nsub0 = half * 8;
    const int w0 = imin(imax(j0 + qsl - 15, 0), 32) - col0;  // this query's window start (local cols)

    // ---- candidate tiles: 30 rows x 3 col-chunks, split over 8 waves ----
    for (int t = wave; t < 90; t += 8) {
        const int r = t / 3, cc = t - r * 3;
        const int crow = si + r;
        const int ccol = col0 + cc * 16 + qsl;
        const unsigned short* bp =
            xfb + (size_t)(b * NPATCH + crow * M1 + imin(ccol, M1 - 1)) * EDIM;
        __builtin_prefetch(bp + 8 * EDIM, 0, 1);   // global_prefetch_b8 hint

        v8f acc = {0.f, 0.f, 0.f, 0.f, 0.f, 0.f, 0.f, 0.f};
#pragma unroll
        for (int kc = 0; kc < 18; ++kc) {
            u16x8 lo = *reinterpret_cast<const u16x8*>(bp + kc * 32 + bkoff);
            u16x8 hi = *reinterpret_cast<const u16x8*>(bp + kc * 32 + bkoff + 16);
            u16x16 m = __builtin_shufflevector(lo, hi, 0,1,2,3,4,5,6,7,8,9,10,11,12,13,14,15);
            union { u16x16 u; v16bf bf; } cv; cv.u = m;
            acc = __builtin_amdgcn_wmma_f32_16x16x32_bf16(
                false, A[kc], false, cv.bf, (short)0, acc, false, false);
        }

        // transpose 16x16 tile through per-wave LDS (same-wave DS ops are in-order)
#pragma unroll
        for (int r8 = 0; r8 < 8; ++r8) dots[wave][r8 + nsub0][qsl] = acc[r8];
        asm volatile("s_wait_dscnt 0" ::: "memory");

#pragma unroll
        for (int jj = 0; jj < 8; ++jj) {
            const int nn  = nsub0 + jj;
            const int lc  = cc * 16 + nn;
            const int col = col0 + lc;
            const float dot = dots[wave][qsl][nn];
            const float sxn = sx[b * NPATCH + crow * M1 + imin(col, M1 - 1)];
            float d = sxn - 2.0f * dot;            // sy[q] is a per-query constant: add later
            const bool valid = (col < M1) && (lc >= w0) && (lc < w0 + WINS);
            d = valid ? d : INFF;
            ins5(d, crow * M1 + col, tp, ip);
        }
        asm volatile("" ::: "memory");
    }

    // ---- dump partial lists, merge per query ----
    const int list = wave * 2 + half;
#pragma unroll
    for (int k = 0; k < KTOP; ++k) { pd[qsl][list][k] = tp[k]; pidx[qsl][list][k] = ip[k]; }
    __syncthreads();

    if (tid < 16) {
        const int q = tid, qj = j0 + q;
        if (qj < M1) {
            float md[KTOP]; int mi[KTOP];
#pragma unroll
            for (int k = 0; k < KTOP; ++k) { md[k] = INFF; mi[k] = 0; }
            for (int l = 0; l < 16; ++l)
                for (int k = 0; k < KTOP; ++k) ins5(pd[q][l][k], pidx[q][l][k], md, mi);

            const int qn = qi * M1 + qj;
            const float syq = sy[b * NPATCH + qn];
#pragma unroll
            for (int k = 0; k < KTOP; ++k) {
                const float score = md[k] + syq + 1e-5f;
                idx_out[(size_t)(b * NPATCH + qn) * KTOP + k] = mi[k];
                sv_ws[(size_t)(b * KTOP + k) * NPATCH + qn] = __expf(-score * 0.1f);
                fi_s[q][k] = mi[k];
            }
        }
    }
    __syncthreads();

    // ---- per-channel L1 sums for selected matches (f32, from originals) ----
    for (int task = tid; task < 16 * KTOP * CC; task += 256) {
        const int c = task & 63;
        const int k = (task >> 6) % KTOP;
        const int q = task / (KTOP * CC);
        const int qj = j0 + q;
        if (qj >= M1) continue;
        const int n  = fi_s[q][k];
        const int ni = n / M1, nj = n - ni * M1;
        const float* yb = ye + (((size_t)b * CC + c) * HH + qi) * WW + qj;
        const float* xb = xe + (((size_t)b * CC + c) * HH + ni) * WW + nj;
        float s = 0.f;
#pragma unroll
        for (int pi2 = 0; pi2 < 3; ++pi2)
#pragma unroll
            for (int pj2 = 0; pj2 < 3; ++pj2)
                s += fabsf(yb[pi2 * WW + pj2] - xb[pi2 * WW + pj2]);
        dsum_ws[(size_t)((b * KTOP + k) * CC + c) * NPATCH + qi * M1 + qj] = s;
    }
}

// ------------------------------------------------------------------
// Kernel 3: score_img = patch2im(exp(-score/10)) as a gather.
// ------------------------------------------------------------------
__global__ void score_img_kernel(const float* __restrict__ sv_ws, float* __restrict__ out)
{
    const int idx = blockIdx.x * blockDim.x + threadIdx.x;
    if (idx >= BATCH * KTOP * 128 * 128) return;
    const int x = idx & 127, y = (idx >> 7) & 127;
    const int k = (idx >> 14) % KTOP;
    const int b = idx / (KTOP * 128 * 128);
    const int ilo = (y >= 4) ? ((y - 4) >> 1) : 0;
    const int ihi = imin(y >> 1, M1 - 1);
    const int jlo = (x >= 4) ? ((x - 4) >> 1) : 0;
    const int jhi = imin(x >> 1, M1 - 1);
    const float* src = sv_ws + (size_t)(b * KTOP + k) * NPATCH;
    float s = 0.f;
    for (int i = ilo; i <= ihi; ++i)
        for (int j = jlo; j <= jhi; ++j)
            s += src[i * M1 + j];
    out[idx] = s;
}

// ------------------------------------------------------------------
// Kernel 4: diff_img = patch2im(dsum) as a gather.
// ------------------------------------------------------------------
__global__ void diff_img_kernel(const float* __restrict__ dsum_ws, float* __restrict__ out)
{
    const int idx = blockIdx.x * blockDim.x + threadIdx.x;
    if (idx >= BATCH * KTOP * CC * 128 * 128) return;
    const int x = idx & 127, y = (idx >> 7) & 127;
    const int ch = (idx >> 14) % (KTOP * CC);     // ch = k*64 + c
    const int b = idx / (KTOP * CC * 128 * 128);
    const int ilo = (y >= 4) ? ((y - 4) >> 1) : 0;
    const int ihi = imin(y >> 1, M1 - 1);
    const int jlo = (x >= 4) ? ((x - 4) >> 1) : 0;
    const int jhi = imin(x >> 1, M1 - 1);
    const float* src = dsum_ws + (size_t)(b * KTOP * CC + ch) * NPATCH;
    float s = 0.f;
    for (int i = ilo; i <= ihi; ++i)
        for (int j = jlo; j <= jhi; ++j)
            s += src[i * M1 + j];
    out[idx] = s;
}

// ------------------------------------------------------------------
extern "C" void kernel_launch(void* const* d_in, const int* in_sizes, int n_in,
                              void* d_out, int out_size, void* d_ws, size_t ws_size,
                              hipStream_t stream)
{
    (void)in_sizes; (void)n_in; (void)out_size; (void)ws_size;
    const float* xe = (const float*)d_in[0];
    const float* ye = (const float*)d_in[1];

    char* ws = (char*)d_ws;
    size_t o = 0;
    auto take = [&](size_t bytes) -> char* {
        char* p = ws + o;
        o = (o + bytes + 255) & ~(size_t)255;
        return p;
    };
    unsigned short* xfb = (unsigned short*)take((size_t)BATCH * NPATCH * EDIM * 2);
    unsigned short* yfb = (unsigned short*)take((size_t)BATCH * NPATCH * EDIM * 2);
    float* sxp  = (float*)take((size_t)BATCH * NPATCH * 4);
    float* syp  = (float*)take((size_t)BATCH * NPATCH * 4);
    float* svp  = (float*)take((size_t)BATCH * KTOP * NPATCH * 4);
    float* dsp  = (float*)take((size_t)BATCH * KTOP * CC * NPATCH * 4);

    // outputs, concatenated in return order: score_img, idx_k, diff_img
    float* score_img = (float*)d_out;
    int*   idx_out   = (int*)((float*)d_out + (size_t)BATCH * KTOP * 128 * 128);
    float* diff_img  = (float*)d_out + (size_t)BATCH * KTOP * 128 * 128
                                      + (size_t)BATCH * NPATCH * KTOP;

    prep_kernel<<<dim3(NPATCH, BATCH, 2), 576, 0, stream>>>(xe, ye, xfb, yfb, sxp, syp);
    nlm_topk_kernel<<<dim3(4, M1, BATCH), 256, 0, stream>>>(xfb, yfb, sxp, syp, xe, ye,
                                                            idx_out, svp, dsp);
    {
        const int n = BATCH * KTOP * 128 * 128;
        score_img_kernel<<<(n + 255) / 256, 256, 0, stream>>>(svp, score_img);
    }
    {
        const int n = BATCH * KTOP * CC * 128 * 128;
        diff_img_kernel<<<(n + 255) / 256, 256, 0, stream>>>(dsp, diff_img);
    }
}
